// TinyGPT_47416438948541
// MI455X (gfx1250) — matrix operand order repaired and checked by reading, not compiled, not run on hardware
//
#include <hip/hip_runtime.h>

// ---------------------------------------------------------------------------
// TinyGPT forward for gfx1250 (MI455X): bf16 WMMA GEMMs (128x128 tiles,
// double-buffered LDS, async-to-LDS staging) + flash attention.
// ---------------------------------------------------------------------------

#define Bsz   2
#define Tlen  1024
#define Dmod  1024
#define Lnum  6
#define Hnum  16
#define HDim  64
#define DFF   4096
#define Vocab 32000
#define Mrows (Bsz * Tlen)   // 2048

typedef __attribute__((ext_vector_type(16))) __bf16 bf16x16;
typedef __attribute__((ext_vector_type(8)))  float  floatx8;

union Frag {
    bf16x16        v;
    unsigned       u32[8];
    unsigned short u16[16];
};

#if __has_builtin(__builtin_amdgcn_global_load_async_to_lds_b32)
#define HAVE_ASYNC_LDS 1
// builtin expects non-const int* in explicit address spaces:
//   arg0: global (AS1) source, arg1: LDS (AS3) dest, arg2: imm offset, arg3: cpol
#define GAS(p) ((__attribute__((address_space(1))) int*)(p))
#define LAS(p) ((__attribute__((address_space(3))) int*)(p))
#endif

__device__ __forceinline__ void wait_async0() {
#if __has_builtin(__builtin_amdgcn_s_wait_asynccnt)
    __builtin_amdgcn_s_wait_asynccnt(0);
#else
    asm volatile("s_wait_asynccnt 0" ::: "memory");
#endif
}

__device__ __forceinline__ unsigned short f2bf(float f) {
    unsigned u = __builtin_bit_cast(unsigned, f);
    unsigned r = u + 0x7FFFu + ((u >> 16) & 1u);
    return (unsigned short)(r >> 16);
}

__device__ __forceinline__ float gelu_exact(float v) {
    return 0.5f * v * (1.0f + erff(v * 0.70710678118654752f));
}

__device__ __forceinline__ floatx8 wmma_bf16(bf16x16 a, bf16x16 b, floatx8 c) {
    return __builtin_amdgcn_wmma_f32_16x16x32_bf16(
        /*neg_a=*/false, a, /*neg_b=*/false, b,
        /*c_mod=*/(short)0, c, /*reuse_a=*/false, /*reuse_b=*/false);
}

// ---------------------------------------------------------------------------
// Embedding: x[m,:] = tok_emb[idx[m],:] + pos_emb[m % T,:]
// ---------------------------------------------------------------------------
__global__ __launch_bounds__(256)
void embed_kernel(const int* __restrict__ idx, const float* __restrict__ tok,
                  const float* __restrict__ pos, float* __restrict__ x) {
    int m = blockIdx.x;
    int t = m % Tlen;
    int id = idx[m];
    const float* tr = tok + (size_t)id * Dmod;
    const float* pr = pos + (size_t)t * Dmod;
    float* xr = x + (size_t)m * Dmod;
    for (int i = threadIdx.x; i < Dmod; i += 256)
        xr[i] = tr[i] + pr[i];
}

// ---------------------------------------------------------------------------
// LayerNorm (f32 in) -> bf16 out (GEMM A operand)
// ---------------------------------------------------------------------------
__global__ __launch_bounds__(256)
void layernorm_kernel(const float* __restrict__ x, const float* __restrict__ w,
                      const float* __restrict__ b, unsigned short* __restrict__ out) {
    int m = blockIdx.x;
    const float* xr = x + (size_t)m * Dmod;
    __shared__ float red[256];
    int tid = threadIdx.x;

    float s = 0.f;
    for (int i = tid; i < Dmod; i += 256) s += xr[i];
    red[tid] = s; __syncthreads();
    for (int o = 128; o > 0; o >>= 1) {
        if (tid < o) red[tid] += red[tid + o];
        __syncthreads();
    }
    float mu = red[0] * (1.0f / Dmod);
    __syncthreads();

    float vs = 0.f;
    for (int i = tid; i < Dmod; i += 256) { float d = xr[i] - mu; vs += d * d; }
    red[tid] = vs; __syncthreads();
    for (int o = 128; o > 0; o >>= 1) {
        if (tid < o) red[tid] += red[tid + o];
        __syncthreads();
    }
    float rstd = rsqrtf(red[0] * (1.0f / Dmod) + 1e-5f);

    unsigned short* orow = out + (size_t)m * Dmod;
    for (int i = tid; i < Dmod; i += 256)
        orow[i] = f2bf((xr[i] - mu) * rstd * w[i] + b[i]);
}

// ---------------------------------------------------------------------------
// GEMM: C[M,N] = A_bf16[M,K] @ B_f32[K,N]  (B converted to bf16 in LDS)
// Block tile 128x128, K-step 32, 256 threads = 8 waves (4 M-groups x 2 N-groups);
// each wave computes 32x64 (2x4 WMMA tiles) -> 8 WMMA per K-step.
// A staged via async-to-LDS (bf16 already), B staged f32->bf16; double-buffered.
// EPI: 0 = store f32
//      1 = +bias, GELU, store bf16
//      2 = +resid, store f32
//      3 = +bias +resid, store f32
// ---------------------------------------------------------------------------
template <int EPI>
__global__ __launch_bounds__(256)
void gemm_bf16_kernel(const unsigned short* __restrict__ A, const float* __restrict__ B,
                      const float* __restrict__ bias, const float* __restrict__ resid,
                      void* __restrict__ Cout, int M, int K, int N) {
    const int LD = 36;                        // padded ushort stride
    __shared__ unsigned short sA[2][128 * 36];  // [row][k]
    __shared__ unsigned short sB[2][128 * 36];  // [n][k] (transposed: K-pairs contiguous)

    int tid  = threadIdx.x;
    int lane = tid & 31;
    int w    = tid >> 5;
    int wm   = w & 3;          // M group: rows wm*32 .. wm*32+31
    int wn   = w >> 2;         // N group: cols wn*64 .. wn*64+63
    int bm   = blockIdx.y * 128;
    int bn   = blockIdx.x * 128;

    floatx8 acc[2][4] = {};

    // ---- tile staging (A: 128x32 bf16 = 2048 dwords; B: 32x128 f32) ----
    auto stage = [&](int buf, int k0) {
        #pragma unroll
        for (int i = 0; i < 8; ++i) {
            int e = tid + i * 256;             // 0..2047 dword pairs
            int row = e >> 4, kp = (e & 15) << 1;
            const unsigned* gsrc = (const unsigned*)&A[(size_t)(bm + row) * K + k0 + kp];
            unsigned short* ldst = &sA[buf][row * LD + kp];
#ifdef HAVE_ASYNC_LDS
            __builtin_amdgcn_global_load_async_to_lds_b32(GAS(gsrc), LAS(ldst), 0, 0);
#else
            *(unsigned*)ldst = *gsrc;
#endif
        }
        #pragma unroll
        for (int i = 0; i < 16; ++i) {
            int e = tid + i * 256;             // 0..4095
            int kk = e >> 7, n = e & 127;
            const float* gb = &B[(size_t)(k0 + kk) * N + bn + n];
            sB[buf][n * LD + kk] = f2bf(*gb);
            if (i < 2 && k0 + 64 < K)
                __builtin_prefetch(gb + (size_t)64 * N, 0, 1);  // next-next panel
        }
    };

    stage(0, 0);
    wait_async0();
    __syncthreads();

    int iters = K >> 5;
    for (int it = 0; it < iters; ++it) {
        int cur = it & 1;
        if (it + 1 < iters) stage(cur ^ 1, (it + 1) << 5);

        // ---- fragments + 8 WMMA from current buffer ----
        Frag a[2];
        #pragma unroll
        for (int mi = 0; mi < 2; ++mi) {
            int arow = wm * 32 + mi * 16 + (lane & 15);
            int kb = (lane >> 4) * 16;
            #pragma unroll
            for (int v = 0; v < 8; ++v) {
                int kk = kb + (v << 1);
                a[mi].u32[v] = *(const unsigned*)&sA[cur][arow * LD + kk];
            }
        }
        #pragma unroll
        for (int t = 0; t < 4; ++t) {
            Frag bf;
            int n16 = wn * 64 + t * 16 + (lane & 15);
            int kb2 = (lane >> 4) * 16;
            #pragma unroll
            for (int v = 0; v < 8; ++v)
                bf.u32[v] = *(const unsigned*)&sB[cur][n16 * LD + kb2 + 2 * v];
            #pragma unroll
            for (int mi = 0; mi < 2; ++mi)
                acc[mi][t] = wmma_bf16(a[mi].v, bf.v, acc[mi][t]);
        }

        wait_async0();
        __syncthreads();
    }

    // ---- epilogue: C layout lane N=l%16, rows r (lanes 0-15) / r+8 (16-31) ----
    #pragma unroll
    for (int mi = 0; mi < 2; ++mi) {
        int rbase = bm + wm * 32 + mi * 16 + ((lane >> 4) * 8);
        #pragma unroll
        for (int t = 0; t < 4; ++t) {
            int col = bn + wn * 64 + t * 16 + (lane & 15);
            #pragma unroll
            for (int r = 0; r < 8; ++r) {
                float v = acc[mi][t][r];
                size_t off = (size_t)(rbase + r) * N + col;
                if (EPI == 0) {
                    ((float*)Cout)[off] = v;
                } else if (EPI == 1) {
                    v = gelu_exact(v + bias[col]);
                    ((unsigned short*)Cout)[off] = f2bf(v);
                } else if (EPI == 2) {
                    ((float*)Cout)[off] = v + resid[off];
                } else {
                    ((float*)Cout)[off] = v + bias[col] + resid[off];
                }
            }
        }
    }
}

// ---------------------------------------------------------------------------
// Flash attention: grid (B*H, T/64), 4 waves/block, wave -> 16 query rows.
// qkv layout [m, 3D]; Q ch = h*64+d, K ch = D + h*64+d, V ch = 2D + h*64+d.
// Output attn (bf16) [m, D].
// ---------------------------------------------------------------------------
__global__ __launch_bounds__(128)
void attention_kernel(const float* __restrict__ qkv, unsigned short* __restrict__ attnOut) {
    const int C3 = 3 * Dmod;
    int bh = blockIdx.x;
    int b  = bh >> 4;
    int hh = bh & 15;
    int qBaseBlk = blockIdx.y * 64;

    int tid  = threadIdx.x;
    int lane = tid & 31;
    int w    = tid >> 5;
    int qBase = qBaseBlk + w * 16;

    __shared__ unsigned short sK[32 * 68];      // [key][ch]  (pairs along ch)
    __shared__ unsigned short sV[64 * 36];      // [ch][key]  (pairs along key)
    __shared__ unsigned short sP[4][16 * 36];   // per-wave P bounce [row][key]

    // ---- Q fragments (two 16x32 A-frags covering 64 channels), bf16 ----
    Frag qf[2];
    {
        int qrow = qBase + (lane & 15);
        const float* qp = qkv + (size_t)(b * Tlen + qrow) * C3 + hh * HDim;
        int kb = (lane >> 4) * 16;
        #pragma unroll
        for (int f = 0; f < 2; ++f)
            #pragma unroll
            for (int v = 0; v < 8; ++v) {
                int c = f * 32 + kb + (v << 1);
                float2 d = *(const float2*)&qp[c];
                qf[f].u16[2 * v]     = f2bf(d.x);
                qf[f].u16[2 * v + 1] = f2bf(d.y);
            }
    }

    floatx8 o[4] = {};
    float mrow[8], lrow[8];
    int rowg[8];
    #pragma unroll
    for (int r = 0; r < 8; ++r) {
        mrow[r] = -1e30f; lrow[r] = 0.f;
        rowg[r] = qBase + ((lane >> 4) * 8) + r;
    }

    const float scale = 0.125f;  // 1/sqrt(64)
    int kEnd = qBaseBlk + 64;    // block-uniform loop; waves mask surplus tiles

    for (int k0 = 0; k0 < kEnd; k0 += 32) {
        // ---- cooperative K/V tile stage: 32 keys x 64 ch ----
        for (int i = 0; i < 16; ++i) {
            int e = tid + i * 128;
            int key = e >> 6, c = e & 63;
            size_t base = (size_t)(b * Tlen + k0 + key) * C3 + hh * HDim + c;
            sK[key * 68 + c]  = f2bf(qkv[base + Dmod]);
            sV[c * 36 + key]  = f2bf(qkv[base + 2 * Dmod]);
        }
        __syncthreads();

        // ---- S = Q @ K^T : two 16x16 tiles ----
        floatx8 s0 = {}, s1 = {};
        #pragma unroll
        for (int f = 0; f < 2; ++f) {
            Frag kb0, kb1;
            int kcol = lane & 15;
            int cb = f * 32 + (lane >> 4) * 16;
            #pragma unroll
            for (int v = 0; v < 8; ++v) {
                kb0.u32[v] = *(const unsigned*)&sK[kcol * 68 + cb + 2 * v];
                kb1.u32[v] = *(const unsigned*)&sK[(16 + kcol) * 68 + cb + 2 * v];
            }
            s0 = wmma_bf16(qf[f].v, kb0.v, s0);
            s1 = wmma_bf16(qf[f].v, kb1.v, s1);
        }

        // ---- online softmax (per-row reductions across 16-lane halves) ----
        float p0[8], p1[8], alpha[8];
        #pragma unroll
        for (int r = 0; r < 8; ++r) {
            int row = rowg[r];
            int c0 = k0 + (lane & 15);
            int c1 = c0 + 16;
            float v0 = (c0 <= row) ? s0[r] * scale : -1e30f;
            float v1 = (c1 <= row) ? s1[r] * scale : -1e30f;
            float mx = fmaxf(v0, v1);
            #pragma unroll
            for (int off = 8; off >= 1; off >>= 1)
                mx = fmaxf(mx, __shfl_xor(mx, off, 32));
            float mnew = fmaxf(mrow[r], mx);
            float a = __expf(mrow[r] - mnew);
            float e0 = (v0 > -1e29f) ? __expf(v0 - mnew) : 0.f;
            float e1 = (v1 > -1e29f) ? __expf(v1 - mnew) : 0.f;
            float es = e0 + e1;
            #pragma unroll
            for (int off = 8; off >= 1; off >>= 1)
                es += __shfl_xor(es, off, 32);
            lrow[r] = lrow[r] * a + es;
            mrow[r] = mnew;
            alpha[r] = a;
            p0[r] = e0; p1[r] = e1;
        }
        #pragma unroll
        for (int t = 0; t < 4; ++t)
            #pragma unroll
            for (int r = 0; r < 8; ++r) o[t][r] *= alpha[r];

        // ---- re-layout P (C-frag) -> A-frag via per-wave LDS bounce ----
        unsigned short* pw = sP[w];
        #pragma unroll
        for (int r = 0; r < 8; ++r) {
            int prow = ((lane >> 4) * 8) + r;
            pw[prow * 36 + (lane & 15)]      = f2bf(p0[r]);
            pw[prow * 36 + 16 + (lane & 15)] = f2bf(p1[r]);
        }
        asm volatile("s_wait_dscnt 0" ::: "memory");  // wave-local LDS RAW

        Frag pa;
        {
            int prow = lane & 15;
            int kb2 = (lane >> 4) * 16;
            #pragma unroll
            for (int v = 0; v < 8; ++v) {
                int kk = kb2 + (v << 1);
                pa.u32[v] = *(const unsigned*)&pw[prow * 36 + kk];
            }
        }

        // ---- O += P @ V (4 tiles over 64 channels) ----
        #pragma unroll
        for (int t = 0; t < 4; ++t) {
            Frag vb;
            int n = t * 16 + (lane & 15);
            int kb3 = (lane >> 4) * 16;
            #pragma unroll
            for (int v = 0; v < 8; ++v)
                vb.u32[v] = *(const unsigned*)&sV[n * 36 + kb3 + 2 * v];
            o[t] = wmma_bf16(pa.v, vb.v, o[t]);
        }
        __syncthreads();
    }

    // ---- finalize: O /= l, store bf16 into attn[m, h*64 + ch] ----
    #pragma unroll
    for (int r = 0; r < 8; ++r) {
        float inv = 1.0f / lrow[r];
        size_t mo = (size_t)(b * Tlen + rowg[r]) * Dmod + hh * HDim;
        #pragma unroll
        for (int t = 0; t < 4; ++t)
            attnOut[mo + t * 16 + (lane & 15)] = f2bf(o[t][r] * inv);
    }
}

// ---------------------------------------------------------------------------
// Host orchestration
// ---------------------------------------------------------------------------
extern "C" void kernel_launch(void* const* d_in, const int* in_sizes, int n_in,
                              void* d_out, int out_size, void* d_ws, size_t ws_size,
                              hipStream_t stream) {
    (void)in_sizes; (void)n_in; (void)out_size; (void)ws_size;

    const int*   idx   = (const int*)d_in[0];
    const float* tok   = (const float*)d_in[1];
    const float* pos   = (const float*)d_in[2];
    const float* ln1w  = (const float*)d_in[3];
    const float* ln1b  = (const float*)d_in[4];
    const float* qkvw  = (const float*)d_in[5];
    const float* projw = (const float*)d_in[6];
    const float* ln2w  = (const float*)d_in[7];
    const float* ln2b  = (const float*)d_in[8];
    const float* fc1w  = (const float*)d_in[9];
    const float* fc1b  = (const float*)d_in[10];
    const float* fc2w  = (const float*)d_in[11];
    const float* fc2b  = (const float*)d_in[12];
    const float* lnfw  = (const float*)d_in[13];
    const float* lnfb  = (const float*)d_in[14];
    const float* headw = (const float*)d_in[15];

    // Workspace layout (bytes):
    //   x    : f32  [2048,1024]   @ 0          (8 MiB)
    //   xn   : bf16 [2048,1024]   @ 8388608    (4 MiB)
    //   qkv  : f32  [2048,3072]   @ 12582912   (24 MiB)
    //   attn : bf16 [2048,1024]   @ 37748736   (4 MiB)
    //   h    : bf16 [2048,4096]   @ 41943040   (16 MiB)
    char* ws = (char*)d_ws;
    float*          x    = (float*)(ws);
    unsigned short* xn   = (unsigned short*)(ws + 8388608);
    float*          qkv  = (float*)(ws + 12582912);
    unsigned short* attn = (unsigned short*)(ws + 37748736);
    unsigned short* hbuf = (unsigned short*)(ws + 41943040);

    embed_kernel<<<Mrows, 256, 0, stream>>>(idx, tok, pos, x);

    for (int l = 0; l < Lnum; ++l) {
        layernorm_kernel<<<Mrows, 256, 0, stream>>>(x, ln1w + l * Dmod, ln1b + l * Dmod, xn);
        gemm_bf16_kernel<0><<<dim3(24, 16), 256, 0, stream>>>(
            xn, qkvw + (size_t)l * Dmod * 3 * Dmod, nullptr, nullptr, qkv,
            Mrows, Dmod, 3 * Dmod);
        attention_kernel<<<dim3(Bsz * Hnum, Tlen / 64), 128, 0, stream>>>(qkv, attn);
        gemm_bf16_kernel<2><<<dim3(8, 16), 256, 0, stream>>>(
            attn, projw + (size_t)l * Dmod * Dmod, nullptr, x, x,
            Mrows, Dmod, Dmod);
        layernorm_kernel<<<Mrows, 256, 0, stream>>>(x, ln2w + l * Dmod, ln2b + l * Dmod, xn);
        gemm_bf16_kernel<1><<<dim3(32, 16), 256, 0, stream>>>(
            xn, fc1w + (size_t)l * Dmod * DFF, fc1b + l * DFF, nullptr, hbuf,
            Mrows, Dmod, DFF);
        gemm_bf16_kernel<3><<<dim3(8, 16), 256, 0, stream>>>(
            hbuf, fc2w + (size_t)l * DFF * Dmod, fc2b + l * Dmod, x, x,
            Mrows, DFF, Dmod);
    }

    layernorm_kernel<<<Mrows, 256, 0, stream>>>(x, lnfw, lnfb, xn);
    gemm_bf16_kernel<0><<<dim3(250, 16), 256, 0, stream>>>(
        xn, headw, nullptr, nullptr, d_out, Mrows, Dmod, Vocab);
}
